// Gen2SingleInputReadout_19911468384644
// MI455X (gfx1250) — matrix operand order repaired and checked by reading, not compiled, not run on hardware
//
#include <hip/hip_runtime.h>
#include <hip/hip_bf16.h>

// ---------------------------------------------------------------------------
// CDNA5 (gfx1250) chunked gated linear attention.
// wave32, WMMA bf16 16x16x32 (f32 accum), async global->LDS staging,
// WMMA operands staged in LDS: A row-major [M][K], B as [K][N]; k in AMD table order.
// ---------------------------------------------------------------------------

typedef __attribute__((ext_vector_type(16))) __bf16 v16bf;
typedef __attribute__((ext_vector_type(8)))  __bf16 v8bf;
typedef __attribute__((ext_vector_type(8)))  float  v8f;

#define EPS 1e-8f
#define USE_ASYNC 1

__device__ __forceinline__ v8f v8f_zero() {
    v8f z;
#pragma unroll
    for (int i = 0; i < 8; ++i) z[i] = 0.f;
    return z;
}

__device__ __forceinline__ v8f wmma_bf16(v16bf a, v16bf b, v8f c) {
    return __builtin_amdgcn_wmma_f32_16x16x32_bf16(
        /*neg_a=*/false, a, /*neg_b=*/false, b,
        /*c_mod=*/(short)0, c, /*reuse_a=*/false, /*reuse_b=*/false);
}

__device__ __forceinline__ v16bf join8(v8bf lo, v8bf hi) {
    return __builtin_shufflevector(lo, hi, 0, 1, 2, 3, 4, 5, 6, 7,
                                   8, 9, 10, 11, 12, 13, 14, 15);
}

// A matrix 16x32 (MxK), source row-major [row][k], row stride `ld` (16B mult).
// Lane L<16: row L, K = kk+{0..7,16..23}; lane L>=16: row L-16, K = kk+{8..15,24..31}.
__device__ __forceinline__ v16bf load_a_rm(const __bf16* s, int ld,
                                           int row0, int kk, int lane) {
    const __bf16* p = s + (row0 + (lane & 15)) * ld + kk + ((lane >> 4) << 3);
    v8bf lo = *(const v8bf*)(p);
    v8bf hi = *(const v8bf*)(p + 16);
    return join8(lo, hi);
}

// B matrix 32x16 (KxN), source stored [k][n], row stride `ld` (16B mult).
// Lane = N column (lane&15); elements = K = kk+{0..7,16..23} (lane<16) or kk+{8..15,24..31}.
__device__ __forceinline__ v16bf load_b_kn(const __bf16* s, int ld,
                                           int col0, int kk, int lane) {
    const __bf16* p = s + (kk + ((lane >> 4) << 3)) * ld + col0 + (lane & 15);
    v16bf b;
#pragma unroll
    for (int e = 0; e < 16; ++e) b[e] = p[(e < 8 ? e : e + 8) * ld];
    return b;
}

// B matrix 32x16 from f32 source stored [k][n] (the transposed state S^T[n][d]).
__device__ __forceinline__ v16bf load_b_kn_f32(const float* s, int ld,
                                               int col0, int kk, int lane) {
    const float* p = s + (kk + ((lane >> 4) << 3)) * ld + col0 + (lane & 15);
    v16bf b;
#pragma unroll
    for (int e = 0; e < 16; ++e) b[e] = (__bf16)p[(e < 8 ? e : e + 8) * ld];
    return b;
}

// 16-byte global -> LDS copy; async (CDNA5 GLOBAL_LOAD_ASYNC_TO_LDS_B128,
// tracked by ASYNCcnt) or synchronous fallback.
__device__ __forceinline__ void copy16(void* lds, const void* g) {
#if USE_ASYNC
    unsigned int off = (unsigned int)(uintptr_t)lds;  // addr[31:0] = LDS offset
    asm volatile("global_load_async_to_lds_b128 %0, %1, off"
                 :: "v"(off), "v"(g) : "memory");
#else
    *(uint4*)lds = *(const uint4*)g;
#endif
}

__device__ __forceinline__ void copy16_fence() {
#if USE_ASYNC
    asm volatile("s_wait_asynccnt 0x0" ::: "memory");
#endif
}

// ---------------------------------------------------------------------------
// Kernel 0a: f32 -> bf16 bulk convert (x -> xb), 8 elements/thread.
// ---------------------------------------------------------------------------
__global__ __launch_bounds__(256)
void cvt_bf16_kernel(const float* __restrict__ in, __bf16* __restrict__ outp,
                     long n8) {
    long i = (long)blockIdx.x * blockDim.x + threadIdx.x;
    if (i >= n8) return;
    const float4* p = (const float4*)in + i * 2;
    float4 a = p[0], b = p[1];
    v8bf o;
    o[0] = (__bf16)a.x; o[1] = (__bf16)a.y; o[2] = (__bf16)a.z; o[3] = (__bf16)a.w;
    o[4] = (__bf16)b.x; o[5] = (__bf16)b.y; o[6] = (__bf16)b.z; o[7] = (__bf16)b.w;
    ((v8bf*)outp)[i] = o;
}

// ---------------------------------------------------------------------------
// Kernel 0b: weight transpose W[N][K] f32 -> Wt[K][N] bf16.
// ---------------------------------------------------------------------------
__global__ __launch_bounds__(256)
void transpose_w_kernel(const float* __restrict__ W, __bf16* __restrict__ Wt,
                        int N, int K) {
    int idx = blockIdx.x * 256 + threadIdx.x;
    if (idx >= N * K) return;
    int k = idx / N, n = idx - k * N;
    Wt[idx] = (__bf16)W[(size_t)n * K + k];
}

// ---------------------------------------------------------------------------
// Kernel 1: projection GEMM  out[m,n] = sum_k xb[m,k] * Wt[k,n] + bias[n]
// Block: 256 threads (8 waves) -> 128(M) x 64(N) tile. All operands bf16.
// mode: 0 = f32 out, 1 = f32 sigmoid out, 2 = bf16 out.
// ---------------------------------------------------------------------------
__global__ __launch_bounds__(256)
void proj_gemm_kernel(const __bf16* __restrict__ Xb, const __bf16* __restrict__ Wt,
                      const float* __restrict__ bias, float* __restrict__ outF,
                      __bf16* __restrict__ outB, int M, int N, int K, int mode) {
    __shared__ __align__(16) __bf16 sX[128 * 48];   // [m][k], ld 48
    __shared__ __align__(16) __bf16 sWt[32 * 72];   // [k][n], ld 72

    const int lane = threadIdx.x & 31;
    const int wave = threadIdx.x >> 5;
    const int m0 = blockIdx.x * 128;
    const int n0 = blockIdx.y * 64;

    v8f acc[4];
#pragma unroll
    for (int i = 0; i < 4; ++i) acc[i] = v8f_zero();

    for (int kk = 0; kk < K; kk += 32) {
        for (int i8 = threadIdx.x; i8 < 512; i8 += 256) {     // 128x32 bf16
            int r = i8 >> 2, c8 = (i8 & 3) << 3;
            copy16(sX + r * 48 + c8, Xb + (size_t)(m0 + r) * K + kk + c8);
        }
        {
            int i8 = threadIdx.x;                              // 32x64 bf16
            if (i8 < 256) {
                int k = i8 >> 3, n8 = (i8 & 7) << 3;
                copy16(sWt + k * 72 + n8, Wt + (size_t)(kk + k) * N + n0 + n8);
            }
        }
        copy16_fence();
        __syncthreads();

        v16bf a = load_a_rm(sX, 48, wave * 16, 0, lane);
#pragma unroll
        for (int nt = 0; nt < 4; ++nt) {
            v16bf b = load_b_kn(sWt, 72, nt * 16, 0, lane);
            acc[nt] = wmma_bf16(a, b, acc[nt]);
        }
        __syncthreads();
    }

#pragma unroll
    for (int nt = 0; nt < 4; ++nt) {
#pragma unroll
        for (int r = 0; r < 8; ++r) {
            int m = m0 + wave * 16 + ((lane >> 4) << 3) + r;
            int n = n0 + nt * 16 + (lane & 15);
            float vo = acc[nt][r] + bias[n];
            if (mode == 1) vo = 1.f / (1.f + __expf(-vo));
            if (mode == 2) outB[(size_t)m * N + n] = (__bf16)vo;
            else           outF[(size_t)m * N + n] = vo;
        }
    }
}

// ---------------------------------------------------------------------------
// Kernel 2: per-chunk gate. One thread per (chunk, b, n).
// qt  natural [t][n] bf16;  ktT, kpT transposed [cb][n][t] bf16; pend f32.
// ---------------------------------------------------------------------------
__global__ __launch_bounds__(64)
void gate_kernel(const float* __restrict__ q, const float* __restrict__ k,
                 const float* __restrict__ alpha, __bf16* __restrict__ qt,
                 __bf16* __restrict__ ktT, __bf16* __restrict__ kpT,
                 float* __restrict__ pend) {
    const int Bb = 4, DK = 64, Cc = 128;
    const int c = blockIdx.x, b = blockIdx.y, n = threadIdx.x;

    size_t base = (((size_t)c * Cc) * Bb + b) * DK + n;
    size_t step = (size_t)Bb * DK;
    size_t tbase = (((size_t)c * Bb + b) * DK + n) * Cc;

    float pe = 1.f;
    size_t idx = base;
    for (int t = 0; t < Cc; ++t) { pe *= fmaxf(alpha[idx], EPS); idx += step; }
    pend[((size_t)c * Bb + b) * DK + n] = pe;

    float p = 1.f;
    idx = base;
    for (int t = 0; t < Cc; ++t) {
        p *= fmaxf(alpha[idx], EPS);
        float ktv = k[idx] / (p + EPS);
        qt[idx]       = (__bf16)(q[idx] * p);
        ktT[tbase + t] = (__bf16)ktv;
        kpT[tbase + t] = (__bf16)(ktv * pe);
        idx += step;
    }
}

// ---------------------------------------------------------------------------
// Kernel 3: sequential chunk scan. Block owns (batch b, 128-row D_V slice).
// State kept TRANSPOSED in LDS: sSt[n][d] (64x128 f32), resident across chunks.
//   A  = tril(Q̃ K̃ᵀ)  : A rm = sQ[t][n],  B kn = sKt[n][s]       (K = n)
//   Y  = A·V + Q̃·Sᵀ  : A rm = sA / sQ,   B kn = sV[t][d] / sSt   (K = s / n)
//   Sᵀ = Sᵀ∘pe + KPᵀ·V: A rm = sKPt[n][t], B kn = sV[t][d]        (K = t)
// ---------------------------------------------------------------------------
#define OFF_SST   0u
#define OFF_SPE   34816u
#define OFF_SQ    35072u
#define OFF_SKT   53504u
#define OFF_SKPT  70912u
#define OFF_SV    88320u
#define OFF_SA    123136u
#define SCAN_SMEM_BYTES 157952u

__global__ __launch_bounds__(256)
void scan_kernel(const __bf16* __restrict__ qt, const __bf16* __restrict__ ktT,
                 const __bf16* __restrict__ kpT, const float* __restrict__ pend,
                 const __bf16* __restrict__ vb, float* __restrict__ out) {
    extern __shared__ __align__(16) char smem[];
    float*  sSt  = (float*)(smem + OFF_SST);    // [64][136] f32  (S^T[n][d])
    float*  sPe  = (float*)(smem + OFF_SPE);    // [64]
    __bf16* sQ   = (__bf16*)(smem + OFF_SQ);    // [128][72]  q̃[t][n]
    __bf16* sKt  = (__bf16*)(smem + OFF_SKT);   // [64][136]  k̃ᵀ[n][s]
    __bf16* sKPt = (__bf16*)(smem + OFF_SKPT);  // [64][136]  (k̃ p_end)ᵀ[n][t]
    __bf16* sV   = (__bf16*)(smem + OFF_SV);    // [128][136] v[t][d]
    __bf16* sA   = (__bf16*)(smem + OFF_SA);    // [128][136] scores[t][s]

    const int lane = threadIdx.x & 31;
    const int wave = threadIdx.x >> 5;
    const int Bb = 4, DK = 64, Cc = 128, DV = 512, NC = 64, DSL = 128;
    const int d0 = blockIdx.x * DSL;
    const int b  = blockIdx.y;

    for (int i = threadIdx.x; i < 64 * 136; i += 256) sSt[i] = 0.f;
    __syncthreads();

    for (int c = 0; c < NC; ++c) {
        size_t tb0  = ((size_t)c * Cc) * Bb + b;          // flat (t,b) row at t=0
        size_t trow = ((size_t)c * Bb + b) * DK;          // [cb][n] row base

        // ---- stage chunk tiles (pure 16B copies; async to LDS) ----
        for (int i8 = threadIdx.x; i8 < 1024; i8 += 256) {           // sQ 128x64
            int t = i8 >> 3, n8 = (i8 & 7) << 3;
            copy16(sQ + t * 72 + n8, qt + (tb0 + (size_t)t * Bb) * DK + n8);
        }
        for (int i8 = threadIdx.x; i8 < 1024; i8 += 256) {           // sKt 64x128
            int n = i8 >> 4, t8 = (i8 & 15) << 3;
            copy16(sKt + n * 136 + t8, ktT + (trow + n) * Cc + t8);
        }
        for (int i8 = threadIdx.x; i8 < 1024; i8 += 256) {           // sKPt 64x128
            int n = i8 >> 4, t8 = (i8 & 15) << 3;
            copy16(sKPt + n * 136 + t8, kpT + (trow + n) * Cc + t8);
        }
        for (int i8 = threadIdx.x; i8 < 2048; i8 += 256) {           // sV 128x128
            int t = i8 >> 4, d8 = (i8 & 15) << 3;
            copy16(sV + t * 136 + d8, vb + (tb0 + (size_t)t * Bb) * DV + d0 + d8);
        }
        copy16_fence();
        if (threadIdx.x < 64)
            sPe[threadIdx.x] = pend[trow + threadIdx.x];
        __syncthreads();

        // ---- phase A: A = tril(Q̃ K̃ᵀ), rows t = wave*16.. ----
        {
            v8f acc[8];
#pragma unroll
            for (int i = 0; i < 8; ++i) acc[i] = v8f_zero();
#pragma unroll
            for (int kk = 0; kk < 64; kk += 32) {
                v16bf a = load_a_rm(sQ, 72, wave * 16, kk, lane);
#pragma unroll
                for (int ct = 0; ct < 8; ++ct) {
                    v16bf bb = load_b_kn(sKt, 136, ct * 16, kk, lane);
                    acc[ct] = wmma_bf16(a, bb, acc[ct]);
                }
            }
#pragma unroll
            for (int ct = 0; ct < 8; ++ct)
#pragma unroll
                for (int r = 0; r < 8; ++r) {
                    int t = wave * 16 + ((lane >> 4) << 3) + r;
                    int s = ct * 16 + (lane & 15);
                    sA[t * 136 + s] = (__bf16)((s <= t) ? acc[ct][r] : 0.f);
                }
        }
        __syncthreads();

        // ---- phase B: Y = A·V + Q̃·Sᵀ (OLD S), rows t = wave*16.. ----
        {
            v8f acc[8];
#pragma unroll
            for (int i = 0; i < 8; ++i) acc[i] = v8f_zero();
#pragma unroll
            for (int kk = 0; kk < 128; kk += 32) {      // intra, K = s
                v16bf a = load_a_rm(sA, 136, wave * 16, kk, lane);
#pragma unroll
                for (int ct = 0; ct < 8; ++ct) {
                    v16bf bb = load_b_kn(sV, 136, ct * 16, kk, lane);
                    acc[ct] = wmma_bf16(a, bb, acc[ct]);
                }
            }
#pragma unroll
            for (int kk = 0; kk < 64; kk += 32) {       // cross, K = n
                v16bf a = load_a_rm(sQ, 72, wave * 16, kk, lane);
#pragma unroll
                for (int ct = 0; ct < 8; ++ct) {
                    v16bf bb = load_b_kn_f32(sSt, 136, ct * 16, kk, lane);
                    acc[ct] = wmma_bf16(a, bb, acc[ct]);
                }
            }
#pragma unroll
            for (int ct = 0; ct < 8; ++ct)
#pragma unroll
                for (int r = 0; r < 8; ++r) {
                    int t = wave * 16 + ((lane >> 4) << 3) + r;
                    int d = ct * 16 + (lane & 15);
                    out[(tb0 + (size_t)t * Bb) * DV + d0 + d] = acc[ct][r];
                }
        }
        __syncthreads();

        // ---- phase C: Sᵀ = Sᵀ∘pe + KPᵀ·V. Wave w: rows (w&3)*16, col half w>>2 ----
        {
            const int r0 = (wave & 3) * 16;
            const int h0 = (wave >> 2) * 64;
            v8f acc[4];
#pragma unroll
            for (int i = 0; i < 4; ++i) acc[i] = v8f_zero();
#pragma unroll
            for (int kk = 0; kk < 128; kk += 32) {      // K = t
                v16bf a = load_a_rm(sKPt, 136, r0, kk, lane);
#pragma unroll
                for (int ct = 0; ct < 4; ++ct) {
                    v16bf bb = load_b_kn(sV, 136, h0 + ct * 16, kk, lane);
                    acc[ct] = wmma_bf16(a, bb, acc[ct]);
                }
            }
            const int rbase = r0 + ((lane >> 4) << 3);
            float per[8];
#pragma unroll
            for (int r = 0; r < 8; ++r) per[r] = sPe[rbase + r];
#pragma unroll
            for (int ct = 0; ct < 4; ++ct)
#pragma unroll
                for (int r = 0; r < 8; ++r) {
                    int n = rbase + r;
                    int d = h0 + ct * 16 + (lane & 15);
                    sSt[n * 136 + d] = sSt[n * 136 + d] * per[r] + acc[ct][r];
                }
        }
        __syncthreads();
    }
}

// ---------------------------------------------------------------------------
// Host launcher
// ---------------------------------------------------------------------------
extern "C" void kernel_launch(void* const* d_in, const int* in_sizes, int n_in,
                              void* d_out, int out_size, void* d_ws, size_t ws_size,
                              hipStream_t stream) {
    const float* x  = (const float*)d_in[0];
    const float* Wv = (const float*)d_in[1];
    const float* bv = (const float*)d_in[2];
    const float* Wk = (const float*)d_in[3];
    const float* bk = (const float*)d_in[4];
    const float* Wq = (const float*)d_in[5];
    const float* bq = (const float*)d_in[6];
    const float* Wa = (const float*)d_in[7];
    const float* ba = (const float*)d_in[8];
    float* out = (float*)d_out;

    const int T = 8192, B = 4, IN = 512, DV = 512, DK = 64, NC = 64;
    const int M = T * B;

    char* ws = (char*)d_ws;
    size_t off = 0;
    auto grab = [&](size_t bytes) -> char* {
        char* p = ws + off;
        off = (off + bytes + 255) & ~(size_t)255;
        return p;
    };
    __bf16* xb  = (__bf16*)grab((size_t)M * IN * 2);
    __bf16* Wvt = (__bf16*)grab((size_t)IN * DV * 2);
    __bf16* Wkt = (__bf16*)grab((size_t)IN * DK * 2);
    __bf16* Wqt = (__bf16*)grab((size_t)IN * DK * 2);
    __bf16* Wat = (__bf16*)grab((size_t)IN * DK * 2);
    float*  kf  = (float*)grab((size_t)M * DK * 4);
    float*  qf  = (float*)grab((size_t)M * DK * 4);
    float*  af  = (float*)grab((size_t)M * DK * 4);
    __bf16* vbf = (__bf16*)grab((size_t)M * DV * 2);
    __bf16* qtb = (__bf16*)grab((size_t)M * DK * 2);
    __bf16* ktT = (__bf16*)grab((size_t)M * DK * 2);
    __bf16* kpT = (__bf16*)grab((size_t)M * DK * 2);
    float*  pe  = (float*)grab((size_t)NC * B * DK * 4);

    dim3 blk(256);

    // 0) bf16 convert of x; bf16 transposed weights.
    long n8 = (long)M * IN / 8;
    cvt_bf16_kernel<<<dim3((unsigned)((n8 + 255) / 256)), blk, 0, stream>>>(x, xb, n8);
    transpose_w_kernel<<<dim3(DV * IN / 256), blk, 0, stream>>>(Wv, Wvt, DV, IN);
    transpose_w_kernel<<<dim3(DK * IN / 256), blk, 0, stream>>>(Wk, Wkt, DK, IN);
    transpose_w_kernel<<<dim3(DK * IN / 256), blk, 0, stream>>>(Wq, Wqt, DK, IN);
    transpose_w_kernel<<<dim3(DK * IN / 256), blk, 0, stream>>>(Wa, Wat, DK, IN);

    // 1) Projections (bf16 WMMA, f32 accumulate).
    proj_gemm_kernel<<<dim3(M / 128, DV / 64), blk, 0, stream>>>(
        xb, Wvt, bv, nullptr, vbf, M, DV, IN, /*mode=*/2);
    proj_gemm_kernel<<<dim3(M / 128, 1), blk, 0, stream>>>(
        xb, Wkt, bk, kf, nullptr, M, DK, IN, /*mode=*/0);
    proj_gemm_kernel<<<dim3(M / 128, 1), blk, 0, stream>>>(
        xb, Wqt, bq, qf, nullptr, M, DK, IN, /*mode=*/0);
    proj_gemm_kernel<<<dim3(M / 128, 1), blk, 0, stream>>>(
        xb, Wat, ba, af, nullptr, M, DK, IN, /*mode=*/1);

    // 2) Per-chunk decay gates (emits k̃ᵀ and (k̃ p_end)ᵀ pre-transposed).
    gate_kernel<<<dim3(NC, B), 64, 0, stream>>>(qf, kf, af, qtb, ktT, kpT, pe);

    // 3) Sequential chunk scan; Sᵀ resident in LDS.
    hipFuncSetAttribute(reinterpret_cast<const void*>(scan_kernel),
                        hipFuncAttributeMaxDynamicSharedMemorySize,
                        (int)SCAN_SMEM_BYTES);
    scan_kernel<<<dim3(DV / 128, B), blk, SCAN_SMEM_BYTES, stream>>>(
        qtb, ktT, kpT, pe, vbf, out);
}
